// CrossAttention_50379966382362
// MI455X (gfx1250) — compile-verified, matrix operand order checked by reading, not run on hardware
//
#include <hip/hip_runtime.h>
#include <math.h>

// ---------------- types ----------------
typedef _Float16 f16;
typedef __attribute__((ext_vector_type(16))) _Float16 v16h;
typedef __attribute__((ext_vector_type(8)))  _Float16 v8h;
typedef __attribute__((ext_vector_type(8)))  float    v8f;

// ---------------- problem dims ----------------
#define DMODEL 512
#define LSEQ   77
#define LP     80      // L padded to multiple of 16
#define LPP    96      // L padded to multiple of 32 (K-dim of P@V)
#define NH     4
#define DH     128
#define BATCH  8
#define TS     12544   // T*S
#define MQ     128     // query rows per workgroup

// ---------------- ws layout (f16 element offsets) ----------------
static const size_t OFF_WQB = 0;                       // Wq as-is [d][j]          262144
static const size_t OFF_WKT = 262144;                  // Wk^T [n][k]              262144
static const size_t OFF_WVT = 524288;                  // Wv^T [n][k]              262144
static const size_t OFF_WOT = 786432;                  // Wo^T [n][k]              262144
static const size_t OFF_KB  = 1048576;                 // K  [b][h][LP][DH]        327680
static const size_t OFF_VT  = 1376256;                 // V^T[b][h][DH][LPP]       393216
static const size_t OFF_MT  = 1769472;                 // M^T[b][h][LP][DMODEL]    1310720
static const size_t OFF_AO  = 3080192;                 // AO [b*TS][DMODEL]        51380224
static const size_t F16_END = 54460416;
static const size_t OFF_BIAS_BYTES = F16_END * 2;      // float [b][LPP]
static const size_t WS_NEEDED = OFF_BIAS_BYTES + (size_t)BATCH * LPP * 4;

// ---------------- wave-level WMMA helpers ----------------
// A operand 16x32 (row-major [M][K]): lane i16 -> row, element e -> K = 16*(e/8) + 8*g + e%8
// B operand 32x16 fed as rows of B^T (row-major [N][K]): identical striping.
__device__ __forceinline__ v16h load_tile(const f16* p) {
  const v8h* q = (const v8h*)p;
  v8h lo = q[0];
  v8h hi = q[2];  // +16 elements
  return __builtin_shufflevector(lo, hi, 0,1,2,3,4,5,6,7,8,9,10,11,12,13,14,15);
}
__device__ __forceinline__ v8f wmma16(v16h a, v16h b, v8f c) {
  return __builtin_amdgcn_wmma_f32_16x16x32_f16(false, a, false, b, (short)0, c, false, false);
}

// ---------------- CDNA5 async global->LDS (ASYNCcnt) ----------------
__device__ __forceinline__ void async_ld16(void* lds_dst, const f16* g_src) {
  // generic pointer to LDS: low 32 bits == LDS byte address
  unsigned loff = (unsigned)(unsigned long long)lds_dst;
  asm volatile("global_load_async_to_lds_b128 %0, %1, off"
               :: "v"(loff), "v"(g_src)
               : "memory");
}
__device__ __forceinline__ void wait_async0() {
  asm volatile("s_wait_asynccnt 0" ::: "memory");
}

union H8 { f16 h[8]; uint4 v; };
union H4 { f16 h[4]; unsigned long long u; };

// =====================================================================
// Kernel 1: convert weights to f16 (Wq plain, Wk/Wv/Wo transposed)
// =====================================================================
__global__ __launch_bounds__(256) void conv_kernel(const float* __restrict__ Wq,
    const float* __restrict__ Wk, const float* __restrict__ Wv, const float* __restrict__ Wo,
    f16* WQB, f16* WKT, f16* WVT, f16* WOT) {
  int i = blockIdx.x * 256 + threadIdx.x;       // 0 .. 262143
  int n = i >> 9, k = i & 511;
  WQB[i] = (f16)Wq[i];
  WKT[i] = (f16)Wk[k * DMODEL + n];
  WVT[i] = (f16)Wv[k * DMODEL + n];
  WOT[i] = (f16)Wo[k * DMODEL + n];
}

// =====================================================================
// Kernel 2: per-batch LayerNorm(text) -> K rows, V^T, mask bias
// =====================================================================
__global__ __launch_bounds__(256) void kv_kernel(const float* __restrict__ text,
    const unsigned char* __restrict__ mask, const float* __restrict__ gamma,
    const float* __restrict__ beta, const f16* __restrict__ WKT, const f16* __restrict__ WVT,
    f16* KB, f16* VT, float* BIAS) {
  extern __shared__ char smem[];
  f16* tn = (f16*)smem;                         // [LP][DMODEL]  81920 B
  int b = blockIdx.x;
  int tid = threadIdx.x, lane = tid & 31, wid = tid >> 5;

  for (int i = tid; i < 3 * DMODEL; i += 256) tn[LSEQ * DMODEL + i] = (f16)0.0f;

  for (int r = wid; r < LSEQ; r += 8) {
    const float* x = text + (size_t)(b * LSEQ + r) * DMODEL;
    float s = 0.f, ss = 0.f, vals[16];
    #pragma unroll
    for (int j = 0; j < 16; ++j) { float v = x[lane + 32 * j]; vals[j] = v; s += v; ss += v * v; }
    #pragma unroll
    for (int m = 16; m >= 1; m >>= 1) { s += __shfl_xor(s, m, 32); ss += __shfl_xor(ss, m, 32); }
    float mu = s * (1.f / 512.f);
    float var = ss * (1.f / 512.f) - mu * mu;
    float rstd = rsqrtf(var + 1e-6f);
    #pragma unroll
    for (int j = 0; j < 16; ++j) {
      int c = lane + 32 * j;
      tn[r * DMODEL + c] = (f16)((vals[j] - mu) * rstd * gamma[c] + beta[c]);
    }
  }
  {
    uint4* vt4 = (uint4*)(VT + (size_t)b * NH * DH * LPP);
    uint4 z = make_uint4(0, 0, 0, 0);
    for (int i = tid; i < NH * DH * LPP / 8; i += 256) vt4[i] = z;
  }
  if (tid < LPP)
    BIAS[b * LPP + tid] = (tid < LSEQ && mask[b * LSEQ + tid] != 0) ? 0.f : -INFINITY;
  __syncthreads();

  int i16 = lane & 15, g = (lane >> 4) & 1;
  for (int which = 0; which < 2; ++which) {
    const f16* WT = which ? WVT : WKT;
    for (int job = wid; job < 80; job += 8) {   // 5 mtiles x 16 npairs
      int m0 = (job / 16) * 16, n0 = (job % 16) * 32;
      v8f acc0 = {}, acc1 = {};
      for (int ks = 0; ks < DMODEL; ks += 32) {
        v16h a  = load_tile(&tn[(m0 + i16) * DMODEL + ks + g * 8]);
        v16h b0 = load_tile(&WT[(size_t)(n0 + i16) * DMODEL + ks + g * 8]);
        v16h b1 = load_tile(&WT[(size_t)(n0 + 16 + i16) * DMODEL + ks + g * 8]);
        acc0 = wmma16(a, b0, acc0);
        acc1 = wmma16(a, b1, acc1);
      }
      #pragma unroll
      for (int half = 0; half < 2; ++half) {
        v8f acc = half ? acc1 : acc0;
        int n = n0 + half * 16 + i16;
        int h = n / DH, kd = n % DH;
        if (which == 0) {                       // K rows: KB[b][h][m][kd]
          f16* kb = KB + ((size_t)(b * NH + h) * LP) * DH + kd;
          #pragma unroll
          for (int r = 0; r < 8; ++r) kb[(m0 + r + 8 * g) * DH] = (f16)acc[r];
        } else {                                // V^T: VT[b][h][kd][m], 8 contiguous
          H8 t;
          #pragma unroll
          for (int r = 0; r < 8; ++r) t.h[r] = (f16)acc[r];
          *(uint4*)(VT + ((size_t)(b * NH + h) * DH + kd) * LPP + m0 + 8 * g) = t.v;
        }
      }
    }
  }
}

// =====================================================================
// Kernel 3: M^T[b][h][l][d] = sum_kd K_h[l][kd] * Wq[d][h*128+kd]
// =====================================================================
__global__ __launch_bounds__(256) void m_kernel(const f16* __restrict__ WQB,
    const f16* __restrict__ KB, f16* MT) {
  int bh = blockIdx.x;                          // 0..31
  __shared__ f16 Ks[LP * DH];                   // 20 KB
  int tid = threadIdx.x, lane = tid & 31, wid = tid >> 5;
  {
    const f16* src = KB + (size_t)bh * LP * DH;
    for (int i = tid; i < LP * DH / 8; i += 256) async_ld16(Ks + i * 8, src + i * 8);
  }
  wait_async0();
  __syncthreads();
  int i16 = lane & 15, g = (lane >> 4) & 1;
  int h = bh & 3;
  const f16* WQh = WQB + h * DH;                // row d: WQB[d*512 + h*128 + k]
  f16* MTb = MT + (size_t)bh * LP * DMODEL;
  for (int job = wid; job < 80; job += 8) {
    int m0 = (job / 16) * 16, n0 = (job % 16) * 32;
    v8f acc0 = {}, acc1 = {};
    #pragma unroll
    for (int ks = 0; ks < DH; ks += 32) {
      v16h a  = load_tile(&Ks[(m0 + i16) * DH + ks + g * 8]);
      v16h b0 = load_tile(&WQh[(size_t)(n0 + i16) * DMODEL + ks + g * 8]);
      v16h b1 = load_tile(&WQh[(size_t)(n0 + 16 + i16) * DMODEL + ks + g * 8]);
      acc0 = wmma16(a, b0, acc0);
      acc1 = wmma16(a, b1, acc1);
    }
    #pragma unroll
    for (int half = 0; half < 2; ++half) {
      v8f acc = half ? acc1 : acc0;
      int n = n0 + half * 16 + i16;
      #pragma unroll
      for (int r = 0; r < 8; ++r) MTb[(m0 + r + 8 * g) * DMODEL + n] = (f16)acc[r];
    }
  }
}

// =====================================================================
// Kernel 4: attention. scores = X @ M_h (+bias, *scale) -> softmax -> P @ V
// =====================================================================
__global__ __launch_bounds__(256) void attn_kernel(const float* __restrict__ visual,
    const f16* __restrict__ MT, const f16* __restrict__ VT, const float* __restrict__ BIAS,
    f16* AO) {
  extern __shared__ char smem[];
  f16*   Xc    = (f16*)(smem);                  // [128][64]  16384 B
  f16*   Ms    = (f16*)(smem + 16384);          // [80][64]   10240 B
  f16*   Ps    = (f16*)(smem + 26624);          // [128][96]  24576 B
  f16*   Vts   = (f16*)(smem + 51200);          // [128][96]  24576 B
  float* Ss    = (float*)(smem + 75776);        // [128][80]  40960 B
  float* biasS = (float*)(smem + 116736);       // [96]
  int qt = blockIdx.x, b = blockIdx.y;
  int q0 = qt * MQ;
  int tid = threadIdx.x, lane = tid & 31, wid = tid >> 5;
  int i16 = lane & 15, g = (lane >> 4) & 1;
  if (tid < LPP) biasS[tid] = BIAS[b * LPP + tid];
  const float scale = 0.088388347648318447f;    // dh^-0.5

  for (int h = 0; h < NH; ++h) {
    // ---- scores: X(128x512) @ M^T rows, K streamed in 64-chunks ----
    v8f s[5] = {};
    const f16* MTbh = MT + (size_t)(b * NH + h) * LP * DMODEL;
    for (int kc = 0; kc < 8; ++kc) {
      __syncthreads();
      if (tid < 160) { // async-stage M^T chunk (already f16)
        int row = tid >> 1, c0 = (tid & 1) * 32;
        const f16* src = MTbh + (size_t)row * DMODEL + kc * 64 + c0;
        f16* dst = &Ms[row * 64 + c0];
        #pragma unroll
        for (int j = 0; j < 4; ++j) async_ld16(dst + j * 8, src + j * 8);
      }
      { // load X chunk, f32 -> f16 (needs conversion: register path)
        int row = tid >> 1, c0 = (tid & 1) * 32;
        const float4* src = (const float4*)(visual + ((size_t)(b * TS + q0 + row)) * DMODEL + kc * 64 + c0);
        f16* dstp = &Xc[row * 64 + c0];
        #pragma unroll
        for (int j = 0; j < 8; ++j) {
          float4 v = src[j];
          H4 t; t.h[0] = (f16)v.x; t.h[1] = (f16)v.y; t.h[2] = (f16)v.z; t.h[3] = (f16)v.w;
          *(unsigned long long*)(dstp + 4 * j) = t.u;
        }
      }
      wait_async0();
      __syncthreads();
      int m0 = wid * 16;
      #pragma unroll
      for (int ks = 0; ks < 64; ks += 32) {
        v16h a = load_tile(&Xc[(m0 + i16) * 64 + ks + g * 8]);
        #pragma unroll
        for (int nt = 0; nt < 5; ++nt) {
          v16h bb = load_tile(&Ms[(nt * 16 + i16) * 64 + ks + g * 8]);
          s[nt] = wmma16(a, bb, s[nt]);
        }
      }
    }
    __syncthreads();
    { // scale + bias -> Ss
      int m0 = wid * 16;
      #pragma unroll
      for (int nt = 0; nt < 5; ++nt) {
        float bv = biasS[nt * 16 + i16];
        #pragma unroll
        for (int r = 0; r < 8; ++r)
          Ss[(m0 + r + 8 * g) * LP + nt * 16 + i16] = s[nt][r] * scale + bv;
      }
    }
    __syncthreads();
    // ---- softmax (threads 0..127) || async-stage V^T (threads 128..255) ----
    if (tid < MQ) {
      float mx = -INFINITY;
      for (int j = 0; j < LP; ++j) mx = fmaxf(mx, Ss[tid * LP + j]);
      float sum = 0.f;
      for (int j = 0; j < LP; ++j) { float e = __expf(Ss[tid * LP + j] - mx); Ss[tid * LP + j] = e; sum += e; }
      float inv = 1.f / sum;
      for (int j = 0; j < LP; ++j) Ps[tid * LPP + j] = (f16)(Ss[tid * LP + j] * inv);
      for (int j = LP; j < LPP; ++j) Ps[tid * LPP + j] = (f16)0.0f;
    } else {
      int t = tid - MQ;
      const f16* src = VT + (size_t)(b * NH + h) * DH * LPP;
      for (int i = t; i < DH * LPP / 8; i += 128) async_ld16(Vts + i * 8, src + i * 8);
    }
    wait_async0();
    __syncthreads();
    // ---- O_h = P(128x96) @ V(96x128) ----
    {
      v8f o[8] = {};
      int m0 = wid * 16;
      #pragma unroll
      for (int ks = 0; ks < LPP; ks += 32) {
        v16h a = load_tile(&Ps[(m0 + i16) * LPP + ks + g * 8]);
        #pragma unroll
        for (int nt = 0; nt < 8; ++nt) {
          v16h bb = load_tile(&Vts[(nt * 16 + i16) * LPP + ks + g * 8]);
          o[nt] = wmma16(a, bb, o[nt]);
        }
      }
      #pragma unroll
      for (int nt = 0; nt < 8; ++nt) {
        int n = h * DH + nt * 16 + i16;
        #pragma unroll
        for (int r = 0; r < 8; ++r)
          AO[((size_t)(b * TS + q0 + m0 + r + 8 * g)) * DMODEL + n] = (f16)o[nt][r];
      }
    }
  }
}

// =====================================================================
// Kernel 5: out = AO @ Wo  (128x128 tiles, double-buffered async LDS pipeline,
//           Wo tile staged in LDS and shared by all 8 waves)
// =====================================================================
__global__ __launch_bounds__(256) void out_kernel(const f16* __restrict__ AO,
    const f16* __restrict__ WOT, float* __restrict__ out) {
  extern __shared__ char smem[];
  // LDS layout: Ac[p] at p*16384, Wc[p] at 32768 + p*16384  (no pointer arrays:
  // computed per-use so nothing becomes a static addrspacecast initializer)
  size_t r0 = (size_t)blockIdx.x * MQ;          // over B*TS rows
  int n0 = blockIdx.y * 128;
  int tid = threadIdx.x, lane = tid & 31, wid = tid >> 5;
  int i16 = lane & 15, g = (lane >> 4) & 1;
  int row = tid >> 1, c0 = (tid & 1) * 32;

  const f16* gA = AO + (r0 + row) * DMODEL + c0;
  const f16* gW = WOT + (size_t)(n0 + row) * DMODEL + c0;

  auto stage = [&](int p, int kc) {
    const f16* ga = gA + kc * 64;
    const f16* gw = gW + kc * 64;
    f16* la = (f16*)(smem + p * 16384) + row * 64 + c0;
    f16* lw = (f16*)(smem + 32768 + p * 16384) + row * 64 + c0;
    #pragma unroll
    for (int j = 0; j < 4; ++j) {
      async_ld16(la + j * 8, ga + j * 8);
      async_ld16(lw + j * 8, gw + j * 8);
    }
  };

  stage(0, 0);
  v8f o[8] = {};
  for (int kc = 0; kc < 8; ++kc) {
    int p = kc & 1;
    wait_async0();
    __syncthreads();
    if (kc < 7) stage(1 - p, kc + 1);   // DMA next chunk while computing this one
    const f16* Ac = (const f16*)(smem + p * 16384);
    const f16* Wc = (const f16*)(smem + 32768 + p * 16384);
    int m0 = wid * 16;
    #pragma unroll
    for (int ks = 0; ks < 64; ks += 32) {
      v16h a = load_tile(&Ac[(m0 + i16) * 64 + ks + g * 8]);
      #pragma unroll
      for (int nt = 0; nt < 8; ++nt) {
        v16h bb = load_tile(&Wc[(nt * 16 + i16) * 64 + ks + g * 8]);
        o[nt] = wmma16(a, bb, o[nt]);
      }
    }
  }
  int m0 = wid * 16;
  #pragma unroll
  for (int nt = 0; nt < 8; ++nt)
    #pragma unroll
    for (int r = 0; r < 8; ++r)
      out[(r0 + m0 + r + 8 * g) * DMODEL + n0 + nt * 16 + i16] = o[nt][r];
}

// =====================================================================
extern "C" void kernel_launch(void* const* d_in, const int* in_sizes, int n_in,
                              void* d_out, int out_size, void* d_ws, size_t ws_size,
                              hipStream_t stream) {
  (void)in_sizes; (void)n_in; (void)out_size;
  const float* visual        = (const float*)d_in[0];
  const float* text          = (const float*)d_in[1];
  const unsigned char* mask  = (const unsigned char*)d_in[2];
  const float* Wq            = (const float*)d_in[3];
  const float* Wk            = (const float*)d_in[4];
  const float* Wv            = (const float*)d_in[5];
  const float* Wo            = (const float*)d_in[6];
  const float* gamma         = (const float*)d_in[7];
  const float* beta          = (const float*)d_in[8];
  float* out                 = (float*)d_out;
  if (ws_size < WS_NEEDED) return;

  f16* wsh = (f16*)d_ws;
  f16* WQB = wsh + OFF_WQB;
  f16* WKT = wsh + OFF_WKT;
  f16* WVT = wsh + OFF_WVT;
  f16* WOT = wsh + OFF_WOT;
  f16* KB  = wsh + OFF_KB;
  f16* VT  = wsh + OFF_VT;
  f16* MT  = wsh + OFF_MT;
  f16* AO  = wsh + OFF_AO;
  float* BIAS = (float*)((char*)d_ws + OFF_BIAS_BYTES);

  conv_kernel<<<1024, 256, 0, stream>>>(Wq, Wk, Wv, Wo, WQB, WKT, WVT, WOT);
  kv_kernel<<<BATCH, 256, LP * DMODEL * 2, stream>>>(text, mask, gamma, beta, WKT, WVT, KB, VT, BIAS);
  m_kernel<<<BATCH * NH, 256, 0, stream>>>(WQB, KB, MT);
  attn_kernel<<<dim3(TS / MQ, BATCH), 256, 117120, stream>>>(visual, MT, VT, BIAS, AO);
  out_kernel<<<dim3(BATCH * TS / MQ, DMODEL / 128), 256, 65536, stream>>>(AO, WOT, out);
}
